// MultiTaskModel_46926812676623
// MI455X (gfx1250) — matrix operand order repaired and checked by reading, not compiled, not run on hardware
//
#include <hip/hip_runtime.h>
#include <hip/hip_bf16.h>

// ---------------------------------------------------------------------------
// Types for CDNA5 WMMA (wave32, 16x16x32 bf16 -> f32)
// ---------------------------------------------------------------------------
typedef __bf16 v16bf  __attribute__((ext_vector_type(16)));
typedef __bf16 bf16x8 __attribute__((ext_vector_type(8)));
typedef float  v8f    __attribute__((ext_vector_type(8)));

static __device__ __forceinline__ __bf16 f2bf(float f) {
  union { float f; unsigned u; } x; x.f = f;
  unsigned r = x.u + 0x7fffu + ((x.u >> 16) & 1u);   // round-to-nearest-even
  union { unsigned short s; __bf16 b; } y;
  y.s = (unsigned short)(r >> 16);
  return y.b;
}

// ---------------------------------------------------------------------------
// Weight pack: f32 W[Nc,K] (row-major, out = x @ W^T) -> bf16 WMMA B-fragments.
// Fragment (kTile 32 x nTile 16): lane = (n%16) + 16*((k%32)>=16),
// half index = k%16  (contiguous k order, same as the A loads: lanes 0-15 hold K 0-15,
// lanes 16-31 hold K 16-31). Zero-pad to Kpad x Npad.
// ---------------------------------------------------------------------------
__global__ void pack_w_kernel(const float* __restrict__ W, int K, int Nc,
                              int Kpad, int Npad, __bf16* __restrict__ out) {
  int tid = blockIdx.x * blockDim.x + threadIdx.x;
  if (tid >= Kpad * Npad) return;
  int k = tid / Npad;
  int n = tid % Npad;
  float val = (k < K && n < Nc) ? W[(size_t)n * K + k] : 0.f;
  int frag = (k >> 5) * (Npad >> 4) + (n >> 4);
  int lane = (n & 15) + (((k & 31) >= 16) ? 16 : 0);
  out[(size_t)frag * 512 + lane * 16 + (k & 15)] = f2bf(val);
}

// Eval-mode BatchNorm folded into per-column affine: scale = g*rsqrt(v+eps),
// shift = b - m*scale.
__global__ void bnprep_kernel(const float* __restrict__ g, const float* __restrict__ b,
                              const float* __restrict__ m, const float* __restrict__ v,
                              float* __restrict__ scale, float* __restrict__ shift, int C) {
  int i = blockIdx.x * blockDim.x + threadIdx.x;
  if (i >= C) return;
  float s = g[i] * rsqrtf(v[i] + 1e-5f);
  scale[i] = s;
  shift[i] = b[i] - m[i] * s;
}

// ---------------------------------------------------------------------------
// GINE edge messages: t[dst] += relu(h[src] + edge_attr @ eW^T + eb)
// One block per edge; tiny K=10 edge-linear on VALU; f32 global atomics.
// ---------------------------------------------------------------------------
__global__ __launch_bounds__(256)
void edge_msg_kernel(const float* __restrict__ h, int ldh,
                     const float* __restrict__ eattr,
                     const float* __restrict__ eW, const float* __restrict__ eb,
                     const int* __restrict__ src, const int* __restrict__ dst,
                     float* __restrict__ t, int C) {
  int e = blockIdx.x;
  int s = src[e], d = dst[e];
  float ea[10];
#pragma unroll
  for (int j = 0; j < 10; ++j) ea[j] = eattr[(size_t)e * 10 + j];
  for (int c = threadIdx.x; c < C; c += blockDim.x) {
    float a = eb[c];
#pragma unroll
    for (int j = 0; j < 10; ++j) a += ea[j] * eW[c * 10 + j];
    float v = h[(size_t)s * ldh + c] + a;
    if (v > 0.f) atomicAdd(&t[(size_t)d * ldh + c], v);
  }
}

// global_add_pool: g[batch[n]] += h[n]   (C = 512)
__global__ __launch_bounds__(256)
void pool_kernel(const float* __restrict__ h, const int* __restrict__ batch,
                 float* __restrict__ g) {
  int n = blockIdx.x;
  int gi = batch[n];
  for (int c = threadIdx.x; c < 512; c += blockDim.x)
    atomicAdd(&g[(size_t)gi * 512 + c], h[(size_t)n * 512 + c]);
}

// ---------------------------------------------------------------------------
// Generic WMMA GEMM: C[M,Nc] = epi(A[M,K] @ W^T + bias)
//   A: f32 row-major (lda), converted to bf16 through double-buffered LDS
//   W: pre-packed bf16 fragments (ksteps x npadTiles x 512 halves)
//   epi mode: 0=bias, 1=bias+relu, 2=bias+relu+BN-affine, 3=bias+sigmoid
//   block = 256 threads (8 waves), tile 128x128, wave tile 32x64
// ---------------------------------------------------------------------------
#define BM 128
#define LDA_S 40  // LDS row stride in bf16 halves (32 + 8 pad, keeps 16B align)

__global__ __launch_bounds__(256)
void gemm_bf16_wmma(const float* __restrict__ A, int lda, int K, int ksteps,
                    const __bf16* __restrict__ Wp, int npadTiles,
                    const float* __restrict__ bias,
                    float* __restrict__ C, int ldc, int colOff, int Nc,
                    int mode, const float* __restrict__ bnScale,
                    const float* __restrict__ bnShift) {
  __shared__ __align__(16) __bf16 As[2][BM * LDA_S];

  const int tid   = threadIdx.x;
  const int lane  = tid & 31;
  const int wid   = tid >> 5;
  const int waveM = wid & 3;   // 32-row strip
  const int waveN = wid >> 2;  // 64-col strip
  const int rowBase = blockIdx.x * BM;
  const bool ldaAligned = (lda & 3) == 0;

  // cooperative A-tile staging: 2 threads per row, 16 f32 each
  const int lrow = tid >> 1;
  const int lk0  = (tid & 1) * 16;
  const float* arowBase = A + (size_t)(rowBase + lrow) * lda + lk0;

  // load one 16-float chunk of the A tile starting at column kt (+lk0)
  auto loadChunk = [&](int kt, float* f) {
    const float* arow = arowBase + kt;
    const int rem = K - (kt + lk0);
    if (rem >= 16) {
      if (ldaAligned) {
        const float4* a4 = (const float4*)arow;     // 16B aligned
        float4 t0 = a4[0], t1 = a4[1], t2 = a4[2], t3 = a4[3];
        f[0] = t0.x;  f[1] = t0.y;  f[2] = t0.z;  f[3] = t0.w;
        f[4] = t1.x;  f[5] = t1.y;  f[6] = t1.z;  f[7] = t1.w;
        f[8] = t2.x;  f[9] = t2.y;  f[10] = t2.z; f[11] = t2.w;
        f[12] = t3.x; f[13] = t3.y; f[14] = t3.z; f[15] = t3.w;
      } else {
#pragma unroll
        for (int i = 0; i < 16; ++i) f[i] = arow[i];  // straight-line, no guards
      }
    } else {
      // ragged K boundary (at most one k-step per GEMM)
#pragma unroll
      for (int i = 0; i < 16; ++i) f[i] = (i < rem) ? arow[i] : 0.f;
    }
  };
  // convert + store chunk into LDS buffer b (two ds_store_b128)
  auto storeChunk = [&](int b, const float* f) {
    bf16x8 h0, h1;
#pragma unroll
    for (int i = 0; i < 8; ++i) { h0[i] = f2bf(f[i]); h1[i] = f2bf(f[8 + i]); }
    __bf16* dp = &As[b][lrow * LDA_S + lk0];
    ((bf16x8*)dp)[0] = h0;
    ((bf16x8*)dp)[1] = h1;
  };

  // A-fragment lane mapping (contiguous k: lane half h holds K 16h..16h+15)
  const int kbase = (lane >> 4) * 16;           // 0 or 16
  const int m0a   = waveM * 32 + (lane & 15);

  v8f acc[2][4];
#pragma unroll
  for (int i = 0; i < 2; ++i)
#pragma unroll
    for (int f = 0; f < 4; ++f)
#pragma unroll
      for (int r = 0; r < 8; ++r) acc[i][f][r] = 0.f;

  // prologue: stage tile 0 into buffer 0
  float stg[16];
  loadChunk(0, stg);
  storeChunk(0, stg);

  for (int ks = 0; ks < ksteps; ++ks) {
    __syncthreads();                            // tile ks visible in As[ks&1]
    const bool more = (ks + 1) < ksteps;
    if (more) loadChunk((ks + 1) * 32, stg);    // issue early; consumed after WMMAs

    const int bsel = ks & 1;
    v16bf afrag[2];
#pragma unroll
    for (int mf = 0; mf < 2; ++mf) {
      const bf16x8* p = (const bf16x8*)&As[bsel][(m0a + mf * 16) * LDA_S + kbase];
      ((bf16x8*)&afrag[mf])[0] = p[0];   // K kbase..kbase+7   -> halves 0-7
      ((bf16x8*)&afrag[mf])[1] = p[1];   // K kbase+8..+15     -> halves 8-15
    }

#pragma unroll
    for (int f = 0; f < 4; ++f) {
      const int nTile = blockIdx.y * 8 + waveN * 4 + f;
      const bf16x8* q =
          (const bf16x8*)(Wp + ((size_t)ks * npadTiles + nTile) * 512 + lane * 16);
      v16bf bfrag;
      ((bf16x8*)&bfrag)[0] = q[0];
      ((bf16x8*)&bfrag)[1] = q[1];
      acc[0][f] = __builtin_amdgcn_wmma_f32_16x16x32_bf16(
          false, afrag[0], false, bfrag, (short)0, acc[0][f], false, false);
      acc[1][f] = __builtin_amdgcn_wmma_f32_16x16x32_bf16(
          false, afrag[1], false, bfrag, (short)0, acc[1][f], false, false);
    }

    if (more) storeChunk((ks + 1) & 1, stg);    // write next tile (other buffer)
  }

  // Epilogue (C/D layout: VGPR r -> row r (lanes 0-15) / r+8 (lanes 16-31))
  const int rhalf = (lane >> 4) * 8;
  const int ncol  = lane & 15;
#pragma unroll
  for (int mf = 0; mf < 2; ++mf) {
#pragma unroll
    for (int f = 0; f < 4; ++f) {
      int col = blockIdx.y * 128 + waveN * 64 + f * 16 + ncol;
      if (col < Nc) {
        float b  = bias[col];
        float sc = 1.f, sh = 0.f;
        if (mode == 2) { sc = bnScale[col]; sh = bnShift[col]; }
#pragma unroll
        for (int r = 0; r < 8; ++r) {
          int row = rowBase + waveM * 32 + mf * 16 + rhalf + r;
          float v = acc[mf][f][r] + b;
          if (mode == 1 || mode == 2) v = v > 0.f ? v : 0.f;
          if (mode == 2) v = v * sc + sh;
          if (mode == 3) v = 1.f / (1.f + __expf(-v));
          C[(size_t)row * ldc + colOff + col] = v;
        }
      }
    }
  }
}

// ---------------------------------------------------------------------------
// Host orchestration
// ---------------------------------------------------------------------------
extern "C" void kernel_launch(void* const* d_in, const int* in_sizes, int n_in,
                              void* d_out, int out_size, void* d_ws, size_t ws_size,
                              hipStream_t stream) {
  (void)in_sizes; (void)n_in; (void)out_size; (void)ws_size;
  const int N = 65536, E = 131072, G = 2048;

  const float* x      = (const float*)d_in[0];
  const float* finger = (const float*)d_in[1];
  const float* eattr  = (const float*)d_in[2];
  const int*   eidx   = (const int*)d_in[3];
  const int*   batch  = (const int*)d_in[4];
  const int*   src    = eidx;
  const int*   dst    = eidx + E;
  auto P = [&](int i) { return (const float*)d_in[i]; };
  // param base index for layer l (1..3): 5 + (l-1)*10
  // +0 eW, +1 eb, +2 w1, +3 b1, +4 w2, +5 b2, +6 bng, +7 bnb, +8 bnm, +9 bnv
  const int FCG_W = 35, FCG_B = 36, FP1_W = 37, FP1_B = 38, FP2_W = 39, FP2_B = 40;
  const int FB_W[4] = {41, 43, 45, 47}, FB_B[4] = {42, 44, 46, 48};

  char* ws = (char*)d_ws;
  size_t off = 0;
  auto alloc = [&](size_t bytes) -> void* {
    void* p = ws + off;
    off = (off + bytes + 255) & ~(size_t)255;
    return p;
  };

  float* buf0 = (float*)alloc((size_t)N * 512 * 4);  // t (residual+agg)
  float* buf1 = (float*)alloc((size_t)N * 512 * 4);  // u (hidden)
  float* buf2 = (float*)alloc((size_t)N * 512 * 4);  // h (layer output)
  float* gbuf = (float*)alloc((size_t)G * 512 * 4);
  float* fpb  = (float*)alloc((size_t)G * 128 * 4);
  float* ybuf = (float*)alloc((size_t)G * 512 * 4);  // concat [g | fp]
  float* z1   = (float*)alloc((size_t)G * 256 * 4);
  float* z2   = (float*)alloc((size_t)G * 128 * 4);
  float* z3   = (float*)alloc((size_t)G * 64 * 4);

  struct PW { const __bf16* p; int ksteps; int npadTiles; };
  auto packW = [&](const float* W, int K, int Nc) -> PW {
    int Kp = (K + 31) & ~31;
    int Np = (Nc + 127) & ~127;
    __bf16* outp = (__bf16*)alloc((size_t)Kp * Np * 2);
    int elems = Kp * Np;
    pack_w_kernel<<<(elems + 255) / 256, 256, 0, stream>>>(W, K, Nc, Kp, Np, outp);
    return {outp, Kp / 32, Np / 16};
  };

  // Pack all weights (runs every call; ~3.5 MB of writes, negligible).
  PW pw_n1w1 = packW(P(7), 78, 256);
  PW pw_n1w2 = packW(P(9), 256, 512);
  PW pw_n2w1 = packW(P(17), 512, 512);
  PW pw_n2w2 = packW(P(19), 512, 512);
  PW pw_n3w1 = packW(P(27), 512, 512);
  PW pw_n3w2 = packW(P(29), 512, 512);
  PW pw_fcg  = packW(P(FCG_W), 512, 256);
  PW pw_fp1  = packW(P(FP1_W), 1489, 128);
  PW pw_fp2  = packW(P(FP2_W), 128, 256);
  PW pw_fb1  = packW(P(FB_W[0]), 512, 256);
  PW pw_fb2  = packW(P(FB_W[1]), 256, 128);
  PW pw_fb3  = packW(P(FB_W[2]), 128, 64);
  PW pw_fb4  = packW(P(FB_W[3]), 64, 2);
  PW pw_n1[2] = {pw_n1w1, pw_n1w2};
  PW pw_n2[2] = {pw_n2w1, pw_n2w2};
  PW pw_n3[2] = {pw_n3w1, pw_n3w2};
  PW* pw_layers[3] = {pw_n1, pw_n2, pw_n3};

  float* bnS[3];
  float* bnF[3];
  for (int l = 0; l < 3; ++l) {
    bnS[l] = (float*)alloc(512 * 4);
    bnF[l] = (float*)alloc(512 * 4);
    int b = 5 + l * 10;
    bnprep_kernel<<<2, 256, 0, stream>>>(P(b + 6), P(b + 7), P(b + 8), P(b + 9),
                                         bnS[l], bnF[l], 512);
  }

  auto gemm = [&](const float* A, int lda, int K, PW w, const float* bias,
                  float* C, int ldc, int colOff, int Nc, int M, int mode,
                  const float* s, const float* f) {
    dim3 grid(M / 128, (Nc + 127) / 128);
    gemm_bf16_wmma<<<grid, 256, 0, stream>>>(A, lda, K, w.ksteps, w.p, w.npadTiles,
                                             bias, C, ldc, colOff, Nc, mode, s, f);
  };

  // ---- 3 GINE layers ----
  const float* h   = x;   // current node features
  int          ldh = 78;
  for (int l = 0; l < 3; ++l) {
    int b = 5 + l * 10;
    int Cin = (l == 0) ? 78 : 512;
    // t = h (residual), then t[dst] += relu(h[src] + edge_lin(edge_attr))
    hipMemcpyAsync(buf0, h, (size_t)N * Cin * 4, hipMemcpyDeviceToDevice, stream);
    edge_msg_kernel<<<E, 256, 0, stream>>>(h, ldh, eattr, P(b + 0), P(b + 1),
                                           src, dst, buf0, Cin);
    int Hmid = (l == 0) ? 256 : 512;
    // u = relu(t @ w1^T + b1)
    gemm(buf0, Cin, Cin, pw_layers[l][0], P(b + 3), buf1, Hmid, 0, Hmid, N, 1,
         nullptr, nullptr);
    // h = bn(relu(u @ w2^T + b2))
    gemm(buf1, Hmid, Hmid, pw_layers[l][1], P(b + 5), buf2, 512, 0, 512, N, 2,
         bnS[l], bnF[l]);
    h = buf2;
    ldh = 512;
  }

  // ---- global add pool + graph head ----
  hipMemsetAsync(gbuf, 0, (size_t)G * 512 * 4, stream);
  pool_kernel<<<N, 256, 0, stream>>>(buf2, batch, gbuf);

  // y[:, 0:256]   = relu(g @ fcg^T + b)
  gemm(gbuf, 512, 512, pw_fcg, P(FCG_B), ybuf, 512, 0, 256, G, 1, nullptr, nullptr);
  // fp = relu(finger @ fp1^T + b); y[:, 256:512] = relu(fp @ fp2^T + b)
  gemm(finger, 1489, 1489, pw_fp1, P(FP1_B), fpb, 128, 0, 128, G, 1, nullptr, nullptr);
  gemm(fpb, 128, 128, pw_fp2, P(FP2_B), ybuf, 512, 256, 256, G, 1, nullptr, nullptr);

  // fb1..fb3 relu, fb4 sigmoid -> d_out [2048, 2]
  gemm(ybuf, 512, 512, pw_fb1, P(FB_B[0]), z1, 256, 0, 256, G, 1, nullptr, nullptr);
  gemm(z1, 256, 256, pw_fb2, P(FB_B[1]), z2, 128, 0, 128, G, 1, nullptr, nullptr);
  gemm(z2, 128, 128, pw_fb3, P(FB_B[2]), z3, 64, 0, 64, G, 1, nullptr, nullptr);
  gemm(z3, 64, 64, pw_fb4, P(FB_B[3]), (float*)d_out, 2, 0, 2, G, 3, nullptr, nullptr);
}
